// DiscretisedBNF_5729486373091
// MI455X (gfx1250) — matrix operand order repaired and checked by reading, not compiled, not run on hardware
//
#include <hip/hip_runtime.h>
#include <hip/hip_bf16.h>

// ---------------------------------------------------------------------------
// DiscretisedBNF loss for MI455X (gfx1250, wave32, WMMA).
// B=256, D=1024, H=2048, K=128.
// GEMM1: [mu|t](256x1025) @ W1(1025x2048)  -> f16 WMMA over K=1024 + rank-1 t row
// GEMM2: h(256x2048)      @ W2(2048x2048)  -> f16 WMMA
// Inner loops register-double-buffer the A/B fragments so v_wmma only waits on
// loads issued one full k-iteration earlier (partial s_wait_loadcnt, not 0).
// Then per-element discretised-CDF (127 erf evals, telescoped) + reduction.
// ---------------------------------------------------------------------------

typedef __attribute__((ext_vector_type(16))) _Float16 v16h;
typedef __attribute__((ext_vector_type(8)))  _Float16 v8h;
typedef __attribute__((ext_vector_type(8)))  float    v8f;

#define LN_SIGMA1  (-3.9120230054281460586f)   /* ln(0.02) */
#define NEG_LN_S1  ( 3.9120230054281460586f)
#define Bdim 256
#define Ddim 1024
#define Hdim 2048
#define N2   2048   /* 2*D */

// ---------------- per-row scalars ----------------
__global__ __launch_bounds__(256)
void bnf_rows(const float* __restrict__ t, float* __restrict__ g,
              float* __restrict__ r, float* __restrict__ w) {
  int b = threadIdx.x;
  float tb = t[b];
  float gamma = 1.0f - __expf(2.0f * tb * LN_SIGMA1);   // 1 - sigma1^(2t)
  g[b] = gamma;
  r[b] = sqrtf((1.0f - gamma) / gamma);
  w[b] = __expf(-2.0f * tb * LN_SIGMA1);                // sigma1^(-2t)
}

// ---------------- mu = gamma*x + gamma*(1-gamma)*noise ----------------
__global__ __launch_bounds__(256)
void bnf_mu(const float* __restrict__ x, const float* __restrict__ noise,
            const float* __restrict__ g, float* __restrict__ muf,
            _Float16* __restrict__ muh) {
  int idx = blockIdx.x * 256 + threadIdx.x;      // 0 .. B*D-1
  int b = idx >> 10;
  float gamma = g[b];
  float m = gamma * x[idx] + gamma * (1.0f - gamma) * noise[idx];
  muf[idx] = m;
  muh[idx] = (_Float16)m;
}

// ---------------- pack W (KxN, f32 row-major) into WMMA B-fragment-major f16 ----
// Fragment (kt,nt): 32 lanes x 16 halves; lane: col = lane&15, kh = lane>>4,
// elem j = W[kt*32 + kh*16 + j][nt*16 + col].
__global__ __launch_bounds__(256)
void bnf_pack(const float* __restrict__ W, _Float16* __restrict__ Wp,
              int K, int N) {
  int idx = blockIdx.x * 256 + threadIdx.x;      // one packed half per thread
  int ntiles = N >> 4;
  int tile = idx >> 9;
  int within = idx & 511;
  int lane = within >> 4;
  int j = within & 15;
  int kt = tile / ntiles;
  int nt = tile - kt * ntiles;
  int c  = lane & 15;
  int kh = lane >> 4;
  int k  = kt * 32 + kh * 16 + j;
  Wp[idx] = (_Float16)W[(size_t)k * N + nt * 16 + c];
}

// ---------------- fragment loaders ----------------
// 16-bit A (16x32) from row-major: lane row = lane&15, kh = lane>>4,
// elems 0..15 = K[kh*16 .. kh*16+15].
__device__ __forceinline__
v16h load_a_frag(const _Float16* __restrict__ A, int lda, int mt, int kt, int lane) {
  int rrow = lane & 15, kh = lane >> 4;
  const _Float16* p = A + (size_t)(mt * 16 + rrow) * lda + kt * 32 + kh * 16;
  v8h lo = *(const v8h*)p;
  v8h hi = *(const v8h*)(p + 8);
  return __builtin_shufflevector(lo, hi, 0,1,2,3,4,5,6,7,8,9,10,11,12,13,14,15);
}

__device__ __forceinline__
v16h load_b_frag(const _Float16* __restrict__ Wp, int ntiles, int kt, int nt, int lane) {
  const _Float16* p = Wp + (((size_t)kt * ntiles + nt) << 9) + (lane << 4);
  __builtin_prefetch(p + ((size_t)ntiles << 10), 0, 1);  // +2 k-tiles ahead
  return *(const v16h*)p;
}

// Double-buffered 16x64-per-wave WMMA mainloop shared by both GEMMs.
#define GEMM_MAINLOOP(Abuf, lda, Wp, NTILES, KT)                               \
  v8f acc0 = {}, acc1 = {}, acc2 = {}, acc3 = {};                              \
  v16h fa  = load_a_frag(Abuf, lda, mt, 0, lane);                              \
  v16h fb0 = load_b_frag(Wp, NTILES, 0, nt0 + 0, lane);                        \
  v16h fb1 = load_b_frag(Wp, NTILES, 0, nt0 + 1, lane);                        \
  v16h fb2 = load_b_frag(Wp, NTILES, 0, nt0 + 2, lane);                        \
  v16h fb3 = load_b_frag(Wp, NTILES, 0, nt0 + 3, lane);                        \
  for (int kt = 0; kt < (KT); ++kt) {                                          \
    int ktn = (kt + 1 < (KT)) ? kt + 1 : kt; /* last iter: harmless reload */  \
    v16h fan = load_a_frag(Abuf, lda, mt, ktn, lane);                          \
    v16h fn0 = load_b_frag(Wp, NTILES, ktn, nt0 + 0, lane);                    \
    v16h fn1 = load_b_frag(Wp, NTILES, ktn, nt0 + 1, lane);                    \
    v16h fn2 = load_b_frag(Wp, NTILES, ktn, nt0 + 2, lane);                    \
    v16h fn3 = load_b_frag(Wp, NTILES, ktn, nt0 + 3, lane);                    \
    acc0 = __builtin_amdgcn_wmma_f32_16x16x32_f16(false, fa, false, fb0,       \
                                                  (short)0, acc0, false, false);\
    acc1 = __builtin_amdgcn_wmma_f32_16x16x32_f16(false, fa, false, fb1,       \
                                                  (short)0, acc1, false, false);\
    acc2 = __builtin_amdgcn_wmma_f32_16x16x32_f16(false, fa, false, fb2,       \
                                                  (short)0, acc2, false, false);\
    acc3 = __builtin_amdgcn_wmma_f32_16x16x32_f16(false, fa, false, fb3,       \
                                                  (short)0, acc3, false, false);\
    fa = fan; fb0 = fn0; fb1 = fn1; fb2 = fn2; fb3 = fn3;                      \
  }

// ---------------- GEMM1: h = leaky(mu @ W1[:1024] + t*W1[1024] + b1) -------
// grid: 128 blocks x 128 thr (4 waves). wave -> (mt, ngroup of 4 n-tiles).
__global__ __launch_bounds__(128)
void bnf_gemm1(const _Float16* __restrict__ muh, const _Float16* __restrict__ W1p,
               const float* __restrict__ W1full, const float* __restrict__ b1,
               const float* __restrict__ tvec, _Float16* __restrict__ h) {
  int lane = threadIdx.x & 31;
  int wid  = blockIdx.x * 4 + (threadIdx.x >> 5);   // 0..511
  int mt = wid & 15;
  int nt0 = (wid >> 4) * 4;                          // ntiles = 128

  GEMM_MAINLOOP(muh, Ddim, W1p, 128, 32)

  int c = lane & 15, kh = lane >> 4;
  v8f accs[4] = {acc0, acc1, acc2, acc3};
  #pragma unroll
  for (int i = 0; i < 4; ++i) {
    int n = (nt0 + i) * 16 + c;
    float bias = b1[n];
    float w1l  = W1full[(size_t)1024 * Hdim + n];    // last row of W1 (t column)
    #pragma unroll
    for (int v = 0; v < 8; ++v) {
      int m = mt * 16 + kh * 8 + v;
      float val = accs[i][v] + bias + tvec[m] * w1l;
      val = (val >= 0.0f) ? val : 0.01f * val;       // leaky relu
      h[(size_t)m * Hdim + n] = (_Float16)val;
    }
  }
}

// ---------------- GEMM2: out = h @ W2 + b2 ----------------
__global__ __launch_bounds__(128)
void bnf_gemm2(const _Float16* __restrict__ h, const _Float16* __restrict__ W2p,
               const float* __restrict__ b2, float* __restrict__ out2) {
  int lane = threadIdx.x & 31;
  int wid  = blockIdx.x * 4 + (threadIdx.x >> 5);   // 0..511
  int mt = wid & 15;
  int nt0 = (wid >> 4) * 4;                          // ntiles = 128

  GEMM_MAINLOOP(h, Hdim, W2p, 128, 64)

  int c = lane & 15, kh = lane >> 4;
  v8f accs[4] = {acc0, acc1, acc2, acc3};
  #pragma unroll
  for (int i = 0; i < 4; ++i) {
    int n = (nt0 + i) * 16 + c;
    float bias = b2[n];
    #pragma unroll
    for (int v = 0; v < 8; ++v) {
      int m = mt * 16 + kh * 8 + v;
      out2[(size_t)m * N2 + n] = accs[i][v] + bias;
    }
  }
}

// ---------------- discretised CDF + loss partials ----------------
// pO = sum_k kc_k * (F(c_k) - F(c_{k-1})), F(c_0=-1)=0 (clamp), upper clamp never
// fires since c_127 = 0.984375 < 1 -> 127 erf evals, telescoped running F.
__global__ __launch_bounds__(256)
void bnf_loss(const float* __restrict__ x, const float* __restrict__ tvec,
              const float* __restrict__ g, const float* __restrict__ rv,
              const float* __restrict__ wv, const float* __restrict__ muf,
              const float* __restrict__ out2, float* __restrict__ partial) {
  __shared__ float s[256];
  int idx = blockIdx.x * 256 + threadIdx.x;       // 0 .. B*D-1
  int b = idx >> 10, d = idx & 1023;
  float tb = tvec[b];
  float gamma = g[b], rr = rv[b];
  float mu_eps = out2[(size_t)b * N2 + d];
  float ln_sig = out2[(size_t)b * N2 + Ddim + d];
  float mu_x = muf[idx] / gamma - rr * mu_eps;
  float sigma_x = rr * __expf(ln_sig);
  if (tb < 1e-10f) { mu_x = 0.0f; sigma_x = 1.0f; }
  float inv = 0.70710678118654752f / sigma_x;     // 1/(sigma*sqrt(2))
  float p = 0.0f, Fprev = 0.0f;
  for (int k = 1; k < 128; ++k) {
    float c  = (float)k * (1.0f / 64.0f) - 1.0f;                 // 2k/K - 1
    float kc = (float)(2 * k - 1) * (1.0f / 128.0f) - 1.0f;      // (2k-1)/K - 1
    float F  = 0.5f * (1.0f + erff((c - mu_x) * inv));
    p += kc * (F - Fprev);
    Fprev = F;
  }
  float se = x[idx] - p;
  s[threadIdx.x] = wv[b] * se * se;
  __syncthreads();
  for (int off = 128; off > 0; off >>= 1) {
    if (threadIdx.x < off) s[threadIdx.x] += s[threadIdx.x + off];
    __syncthreads();
  }
  if (threadIdx.x == 0) partial[blockIdx.x] = s[0];
}

__global__ __launch_bounds__(256)
void bnf_finalize(const float* __restrict__ partial, float* __restrict__ out) {
  __shared__ float s[256];
  float acc = 0.0f;
  for (int i = threadIdx.x; i < 1024; i += 256) acc += partial[i];
  s[threadIdx.x] = acc;
  __syncthreads();
  for (int off = 128; off > 0; off >>= 1) {
    if (threadIdx.x < off) s[threadIdx.x] += s[threadIdx.x + off];
    __syncthreads();
  }
  if (threadIdx.x == 0)
    out[0] = NEG_LN_S1 * s[0] * (1.0f / (float)(Bdim * Ddim));
}

// ---------------- launch ----------------
extern "C" void kernel_launch(void* const* d_in, const int* in_sizes, int n_in,
                              void* d_out, int out_size, void* d_ws, size_t ws_size,
                              hipStream_t stream) {
  const float* x     = (const float*)d_in[0];   // (256,1024)
  const float* tvec  = (const float*)d_in[1];   // (256,1)
  const float* noise = (const float*)d_in[2];   // (256,1024)
  const float* W1    = (const float*)d_in[3];   // (1025,2048)
  const float* b1    = (const float*)d_in[4];   // (2048,)
  const float* W2    = (const float*)d_in[5];   // (2048,2048)
  const float* b2    = (const float*)d_in[6];   // (2048,)

  char* ws = (char*)d_ws;
  size_t off = 0;
  float*    g    = (float*)(ws + off);  off += 256 * 4;
  float*    rv   = (float*)(ws + off);  off += 256 * 4;
  float*    wv   = (float*)(ws + off);  off += 256 * 4;
  off = (off + 255) & ~(size_t)255;
  float*    muf  = (float*)(ws + off);     off += (size_t)Bdim * Ddim * 4;   // 1 MB
  _Float16* muh  = (_Float16*)(ws + off);  off += (size_t)Bdim * Ddim * 2;   // 0.5 MB
  _Float16* W1p  = (_Float16*)(ws + off);  off += (size_t)1024 * Hdim * 2;   // 4 MB
  _Float16* W2p  = (_Float16*)(ws + off);  off += (size_t)Hdim * N2 * 2;     // 8 MB
  _Float16* hbuf = (_Float16*)(ws + off);  off += (size_t)Bdim * Hdim * 2;   // 1 MB
  float*    out2 = (float*)(ws + off);     off += (size_t)Bdim * N2 * 4;     // 2 MB
  float*    part = (float*)(ws + off);     off += 1024 * 4;

  bnf_rows<<<1, 256, 0, stream>>>(tvec, g, rv, wv);
  bnf_mu<<<(Bdim * Ddim) / 256, 256, 0, stream>>>(x, noise, g, muf, muh);
  bnf_pack<<<(1024 * Hdim) / 256, 256, 0, stream>>>(W1, W1p, 1024, Hdim);
  bnf_pack<<<(Hdim * N2) / 256, 256, 0, stream>>>(W2, W2p, Hdim, N2);
  bnf_gemm1<<<128, 128, 0, stream>>>(muh, W1p, W1, b1, tvec, hbuf);
  bnf_gemm2<<<128, 128, 0, stream>>>(hbuf, W2p, b2, out2);
  bnf_loss<<<(Bdim * Ddim) / 256, 256, 0, stream>>>(x, tvec, g, rv, wv, muf, out2, part);
  bnf_finalize<<<1, 256, 0, stream>>>(part, (float*)d_out);
}
